// Res101_SFCN_42417097016966
// MI455X (gfx1250) — matrix operand order repaired and checked by reading, not compiled, not run on hardware
//
#include <hip/hip_runtime.h>

typedef unsigned short u16;
typedef __attribute__((ext_vector_type(16))) __bf16 v16bf;
typedef __attribute__((ext_vector_type(8)))  float  v8f;
typedef __attribute__((ext_vector_type(8)))  unsigned short u16x8;
typedef __attribute__((ext_vector_type(16))) unsigned short u16x16;

__device__ inline float bf2f(u16 u) {
    return __builtin_bit_cast(float, (unsigned int)u << 16);
}
__device__ inline u16 f2bf(float f) {
    unsigned int u = __builtin_bit_cast(unsigned int, f);
    u += 0x7FFFu + ((u >> 16) & 1u);   // round-to-nearest-even
    return (u16)(u >> 16);
}
__device__ inline __bf16 bfbits(u16 u) { return __builtin_bit_cast(__bf16, u); }

__device__ inline v16bf packA(u16x8 q0, u16x8 q1) {
    u16x16 qq = __builtin_shufflevector(q0, q1, 0, 1, 2, 3, 4, 5, 6, 7,
                                        8, 9, 10, 11, 12, 13, 14, 15);
    return __builtin_bit_cast(v16bf, qq);
}

// ---------------- elementwise helpers ----------------

__global__ __launch_bounds__(256) void k_cvt_bf16(const float* __restrict__ s,
                                                  u16* __restrict__ d, int n) {
    int i = blockIdx.x * 256 + threadIdx.x;
    if (i < n) d[i] = f2bf(s[i]);
}

// convert weight [Cout,K] f32 -> [Cout,K32] bf16 with zero padding (K32 = K rounded to 32)
__global__ __launch_bounds__(256) void k_cvt_pad(const float* __restrict__ s,
                                                 u16* __restrict__ d,
                                                 int Cout, int K, int K32) {
    int i = blockIdx.x * 256 + threadIdx.x;
    int total = Cout * K32;
    if (i >= total) return;
    int m = i / K32;
    int k = i - m * K32;
    d[i] = (k < K) ? f2bf(s[(size_t)m * K + k]) : (u16)0;
}

__global__ __launch_bounds__(256) void k_fold_bn(const float* __restrict__ g,
                                                 const float* __restrict__ b,
                                                 const float* __restrict__ m,
                                                 const float* __restrict__ v,
                                                 float* __restrict__ sc,
                                                 float* __restrict__ bi, int C) {
    int c = blockIdx.x * 256 + threadIdx.x;
    if (c < C) {
        float s = g[c] / sqrtf(v[c] + 1e-5f);
        sc[c] = s;
        bi[c] = b[c] - m[c] * s;
    }
}

__global__ __launch_bounds__(256) void k_bias_only(const float* __restrict__ b,
                                                   float* __restrict__ sc,
                                                   float* __restrict__ bi, int C) {
    int c = blockIdx.x * 256 + threadIdx.x;
    if (c < C) { sc[c] = 1.0f; bi[c] = b[c]; }
}

__global__ __launch_bounds__(256) void k_maxpool(const u16* __restrict__ in,
                                                 u16* __restrict__ out,
                                                 int C, int Hin, int Win,
                                                 int Hout, int Wout) {
    int i = blockIdx.x * 256 + threadIdx.x;
    int total = C * Hout * Wout;
    if (i >= total) return;
    int c = i / (Hout * Wout);
    int rem = i - c * (Hout * Wout);
    int oy = rem / Wout, ox = rem - oy * Wout;
    float mx = -3.4e38f;
    for (int dy = 0; dy < 3; ++dy)
        for (int dx = 0; dx < 3; ++dx) {
            int y = oy * 2 - 1 + dy, x = ox * 2 - 1 + dx;
            if ((unsigned)y < (unsigned)Hin && (unsigned)x < (unsigned)Win)
                mx = fmaxf(mx, bf2f(in[(size_t)c * Hin * Win + y * Win + x]));
        }
    out[i] = f2bf(mx);
}

// ---------------- implicit-GEMM conv via WMMA bf16 ----------------
// D = W[Cout x K32(padded)] * im2col(in)[K x Npix], fused scale/bias(+res)(+relu).
// Block(128) = 4 waves: 32 cout x 256 pixels. Per K-step the block async-copies a
// shared 32x32 weight slab into LDS (global_load_async_to_lds_b128, one per thread),
// each wave reads 2 A fragments back via ds_load_b128 and runs 8 WMMAs (2M x 4N),
// so every B fragment is reused by 2 WMMAs and the A slab by all 4 waves.
__global__ __launch_bounds__(128) void k_conv_wmma(
    const u16* __restrict__ in, const u16* __restrict__ wt,
    const float* __restrict__ sc, const float* __restrict__ bi,
    const u16* __restrict__ res, u16* __restrict__ out,
    int Cin, int Hin, int Win, int Cout, int Hout, int Wout,
    int KH, int KW, int stride, int pad, int dil, int relu, int K32)
{
    __shared__ u16 smA[32 * 32];            // 2 KB weight slab

    const int tid  = threadIdx.x;
    const int lane = tid & 31;
    const int wave = tid >> 5;
    const int Npix = Hout * Wout;
    const int m0 = blockIdx.x * 32;
    const int n0 = (blockIdx.y * 4 + wave) * 64;
    if (n0 >= Npix) return;                 // wave-uniform (Npix % 256 == 0 here)

    const int KHW  = KH * KW;
    const int K    = Cin * KHW;
    const int HWin = Hin * Win;
    const bool fast1x1 = (KHW == 1) && (stride == 1) && (pad == 0);

    // A layout (bf16 16x32): lanes 0-15 M=lane, K=0..15;
    //                        lanes 16-31 M=lane-16, K=16..31
    const int mrow = lane & 15;
    const int kbA  = (lane < 16) ? 0 : 16;
    // B layout (bf16 32x16): lanes 0-15 N=lane K=0..15; lanes 16-31 N=lane-16 K=16..31
    const int ncol = lane & 15;
    const int kbB  = (lane < 16) ? 0 : 16;

    int pixj[4], iy0[4], ix0[4];
#pragma unroll
    for (int j = 0; j < 4; ++j) {
        int p = n0 + j * 16 + ncol;
        if (p >= Npix) p = Npix - 1;        // clamp; store is guarded
        pixj[j] = p;
        int oy = p / Wout, ox = p - oy * Wout;
        iy0[j] = oy * stride - pad;
        ix0[j] = ox * stride - pad;
    }

    // cooperative A-slab loader: thread -> (row = tid/4, 16B chunk = tid%4)
    const int lrow = tid >> 2;
    const int lchk = (tid & 3) * 8;         // u16 offset within 32-element row
    int lr = m0 + lrow; if (lr >= Cout) lr = Cout - 1;
    const u16* gsrc = wt + (size_t)lr * (size_t)K32 + lchk;   // 16B-aligned
    const unsigned ldsoff = (unsigned)(uintptr_t)(&smA[lrow * 32 + lchk]);

    v8f acc[2][4] = {};
    for (int k0 = 0; k0 < K32; k0 += 32) {
        // async copy 32x32 bf16 weight slab to LDS (one b128 per thread)
        unsigned long long ga = (unsigned long long)(uintptr_t)(gsrc + k0);
        asm volatile("global_load_async_to_lds_b128 %0, %1, off"
                     :: "v"(ldsoff), "v"(ga) : "memory");
        asm volatile("s_wait_asynccnt 0x0" ::: "memory");
        __syncthreads();

        __builtin_prefetch(gsrc + k0 + 64, 0, 1);   // slab two K-steps ahead into cache

        // A fragments from LDS (ds_load_b128, 16B-aligned)
        u16x8 a00 = *(const u16x8*)(&smA[mrow * 32 + kbA]);
        u16x8 a01 = *(const u16x8*)(&smA[mrow * 32 + kbA + 8]);
        u16x8 a10 = *(const u16x8*)(&smA[(16 + mrow) * 32 + kbA]);
        u16x8 a11 = *(const u16x8*)(&smA[(16 + mrow) * 32 + kbA + 8]);
        v16bf A0 = packA(a00, a01);
        v16bf A1 = packA(a10, a11);

        const int kb = k0 + kbB;
#pragma unroll
        for (int j = 0; j < 4; ++j) {
            v16bf B;
            if (fast1x1) {
                const u16* bp = in + (size_t)kb * HWin + pixj[j];
#pragma unroll
                for (int e = 0; e < 16; ++e) {
                    u16 bv = (kb + e < K) ? bp[(size_t)e * HWin] : (u16)0;
                    B[e] = bfbits(bv);
                }
            } else {
                int ci  = kb / KHW;
                int rem = kb - ci * KHW;
                int kh  = rem / KW;
                int kw  = rem - kh * KW;
                int iy  = iy0[j] + kh * dil;
                int ix  = ix0[j] + kw * dil;
#pragma unroll
                for (int e = 0; e < 16; ++e) {
                    u16 bv = 0;
                    if (ci < Cin && (unsigned)iy < (unsigned)Hin &&
                        (unsigned)ix < (unsigned)Win)
                        bv = in[(size_t)ci * HWin + iy * Win + ix];
                    B[e] = bfbits(bv);
                    ++kw; ix += dil;                 // carry-increment k->(ci,kh,kw)
                    if (kw == KW) {
                        kw = 0; ix = ix0[j];
                        ++kh; iy += dil;
                        if (kh == KH) { kh = 0; iy = iy0[j]; ++ci; }
                    }
                }
            }
            acc[0][j] = __builtin_amdgcn_wmma_f32_16x16x32_bf16(
                false, A0, false, B, (short)0, acc[0][j], false, false);
            acc[1][j] = __builtin_amdgcn_wmma_f32_16x16x32_bf16(
                false, A1, false, B, (short)0, acc[1][j], false, false);
        }
        __syncthreads();    // slab consumed before next overwrite
    }

    // D layout: VGPR r -> M = r (+8 for lanes 16-31), N = lane&15
    const int moff = (lane >= 16) ? 8 : 0;
#pragma unroll
    for (int t = 0; t < 2; ++t) {
        const int mt = m0 + t * 16;
#pragma unroll
        for (int j = 0; j < 4; ++j) {
            int n = n0 + j * 16 + (lane & 15);
            if (n >= Npix) continue;
#pragma unroll
            for (int r = 0; r < 8; ++r) {
                int m = mt + moff + r;
                if (m < Cout) {
                    float v = acc[t][j][r] * sc[m] + bi[m];
                    if (res) v += bf2f(res[(size_t)m * Npix + n]);
                    if (relu) v = fmaxf(v, 0.0f);
                    out[(size_t)m * Npix + n] = f2bf(v);
                }
            }
        }
    }
}

// ---------------- recurrent scan step (convDU / convLR) ----------------
// cur[c,n] = relu(sum_{ci,t} W[c,ci,t]*prev[ci,n+t-4] + bias[c]) + cur[c,n]
// GEMM 64 x 48, K=576. grid(4); one wave covers all 3 n-subtiles (A reuse x3).
__global__ __launch_bounds__(32) void k_scan_step(
    const u16* __restrict__ prev, u16* __restrict__ cur,
    const u16* __restrict__ wt, const float* __restrict__ bias,
    int estride, int cstride, int len)
{
    const int lane = threadIdx.x;
    const int m0 = blockIdx.x * 16;
    const int K = 576;
    const int mrow = lane & 15, kbA = (lane < 16) ? 0 : 16;
    const int ncol = lane & 15, kbB = (lane < 16) ? 0 : 16;
    const u16* wrow = wt + (m0 + mrow) * K;

    v8f acc[3] = {};
    for (int k0 = 0; k0 < K; k0 += 32) {
        u16x8 q0 = *reinterpret_cast<const u16x8*>(wrow + k0 + kbA);
        u16x8 q1 = *reinterpret_cast<const u16x8*>(wrow + k0 + kbA + 8);
        v16bf A = packA(q0, q1);

        const int kb = k0 + kbB;
#pragma unroll
        for (int j = 0; j < 3; ++j) {
            const int n = j * 16 + ncol;
            int ci = kb / 9;
            int tt = kb - ci * 9;
            int p  = n + tt - 4;
            v16bf B;
#pragma unroll
            for (int e = 0; e < 16; ++e) {
                u16 bv = 0;
                if ((unsigned)p < (unsigned)len)
                    bv = prev[ci * cstride + p * estride];
                B[e] = bfbits(bv);
                ++tt; ++p;
                if (tt == 9) { tt = 0; p = n - 4; ++ci; }
            }
            acc[j] = __builtin_amdgcn_wmma_f32_16x16x32_bf16(
                false, A, false, B, (short)0, acc[j], false, false);
        }
    }

    const int moff = (lane >= 16) ? 8 : 0;
#pragma unroll
    for (int j = 0; j < 3; ++j) {
        int n = j * 16 + ncol;
#pragma unroll
        for (int r = 0; r < 8; ++r) {
            int m = m0 + moff + r;
            int idx = m * cstride + n * estride;
            float v = fmaxf(acc[j][r] + bias[m], 0.0f) + bf2f(cur[idx]);
            cur[idx] = f2bf(v);
        }
    }
}

// ---------------- 1x1 out conv + relu + x8 nearest upsample ----------------
__global__ __launch_bounds__(256) void k_out_up(const u16* __restrict__ act,
                                                const float* __restrict__ w,
                                                const float* __restrict__ b,
                                                float* __restrict__ out) {
    int i = blockIdx.x * 256 + threadIdx.x;
    if (i >= 48 * 48) return;
    int oy = i / 48, ox = i - oy * 48;
    float s = 0.f;
    for (int c = 0; c < 64; ++c) s += bf2f(act[c * 2304 + i]) * w[c];
    s = fmaxf(s + b[0], 0.f);
    int by = oy * 8, bx = ox * 8;
    for (int dy = 0; dy < 8; ++dy) {
        float* row = out + (size_t)(by + dy) * 384 + bx;
        for (int dx = 0; dx < 8; ++dx) row[dx] = s;
    }
}

// ---------------- host orchestration ----------------

extern "C" void kernel_launch(void* const* d_in, const int* in_sizes, int n_in,
                              void* d_out, int out_size, void* d_ws, size_t ws_size,
                              hipStream_t stream)
{
    (void)in_sizes; (void)n_in; (void)out_size; (void)ws_size;
    const float* const* P = reinterpret_cast<const float* const*>(d_in);
    int cur = 0;
    auto next = [&]() -> const float* { return P[cur++]; };

    // workspace partition
    size_t off = 0;
    auto wsAlloc = [&](size_t bytes) -> void* {
        off = (off + 255) & ~(size_t)255;
        void* p = (char*)d_ws + off;
        off += bytes;
        return p;
    };
    const size_t WMAX = 4800000;   // largest padded weight (512*1024*9 = 4,718,592)
    u16*   WA = (u16*)wsAlloc(WMAX * 2);
    float* SC = (float*)wsAlloc(1024 * 4);
    float* BI = (float*)wsAlloc(1024 * 4);
    const size_t ABUF = 2400000;   // largest activation (2,359,296)
    u16* buf[5];
    for (int i = 0; i < 5; ++i) buf[i] = (u16*)wsAlloc(ABUF * 2);
    u16* duW = (u16*)wsAlloc(36864 * 2);
    u16* lrW = (u16*)wsAlloc(36864 * 2);

    auto cvt = [&](const float* s, u16* d, int n) {
        k_cvt_bf16<<<dim3((n + 255) / 256), dim3(256), 0, stream>>>(s, d, n);
    };
    auto cvtW = [&](const float* s, int Cout_, int K_) -> int {
        int K32 = (K_ + 31) & ~31;
        int total = Cout_ * K32;
        k_cvt_pad<<<dim3((total + 255) / 256), dim3(256), 0, stream>>>(
            s, WA, Cout_, K_, K32);
        return K32;
    };
    auto nextBN = [&](const float** p) {
        p[0] = next(); p[1] = next(); p[2] = next(); p[3] = next();
    };
    auto fold = [&](const float* const* bn, int C) {
        k_fold_bn<<<dim3((C + 255) / 256), dim3(256), 0, stream>>>(
            bn[0], bn[1], bn[2], bn[3], SC, BI, C);
    };
    auto biasOnly = [&](const float* b, int C) {
        k_bias_only<<<dim3((C + 255) / 256), dim3(256), 0, stream>>>(b, SC, BI, C);
    };
    auto conv = [&](const u16* in, const u16* w, const u16* res, u16* out,
                    int Cin, int Hin, int Win, int Cout, int KH, int KW,
                    int stride, int pad, int dil, int relu, int K32,
                    int& Hout, int& Wout) {
        Hout = (Hin + 2 * pad - dil * (KH - 1) - 1) / stride + 1;
        Wout = (Win + 2 * pad - dil * (KW - 1) - 1) / stride + 1;
        int Npix = Hout * Wout;
        dim3 g((Cout + 31) / 32, (Npix + 255) / 256);
        k_conv_wmma<<<g, dim3(128), 0, stream>>>(in, w, SC, BI, res, out,
            Cin, Hin, Win, Cout, Hout, Wout, KH, KW, stride, pad, dil, relu, K32);
    };

    // --- input + conv1/bn1/relu + maxpool ---
    const float* x = next();                       // 'x' first in dict order
    cvt(x, buf[0], 3 * 384 * 384);
    const float* wc1 = next();
    const float* bn1p[4]; nextBN(bn1p);
    int K32 = cvtW(wc1, 64, 3 * 49);
    fold(bn1p, 64);
    int H, W;
    conv(buf[0], WA, nullptr, buf[1], 3, 384, 384, 64, 7, 7, 2, 3, 1, 1, K32, H, W);
    {
        int total = 64 * 96 * 96;
        k_maxpool<<<dim3((total + 255) / 256), dim3(256), 0, stream>>>(
            buf[1], buf[2], 64, 192, 192, 96, 96);
    }
    int curB = 2;
    int C = 64; H = 96; W = 96;

    // --- bottleneck ---
    auto bottleneck = [&](int planes, int stride, bool down) {
        // flat param order: c1,b1,c2,b2,c3,b3[,dc,dbn]
        const float* pc1 = next(); const float* b1p[4]; nextBN(b1p);
        const float* pc2 = next(); const float* b2p[4]; nextBN(b2p);
        const float* pc3 = next(); const float* b3p[4]; nextBN(b3p);
        const float* pdc = nullptr;
        const float* bdp[4] = {nullptr, nullptr, nullptr, nullptr};
        if (down) { pdc = next(); nextBN(bdp); }

        int picks[4], np = 0;
        for (int i = 0; i < 5 && np < 4; ++i) if (i != curB) picks[np++] = i;
        u16* t1 = buf[picks[0]];
        u16* t2 = buf[picks[1]];
        u16* tD = buf[picks[2]];
        u16* tY = buf[picks[3]];

        int h1, w1, h2, w2, hd, wd, h3, w3;
        int k32 = cvtW(pc1, planes, C);  fold(b1p, planes);
        conv(buf[curB], WA, nullptr, t1, C, H, W, planes, 1, 1, 1, 0, 1, 1, k32, h1, w1);
        k32 = cvtW(pc2, planes, planes * 9);  fold(b2p, planes);
        conv(t1, WA, nullptr, t2, planes, h1, w1, planes, 3, 3, stride, 1, 1, 1, k32, h2, w2);

        const u16* res = buf[curB];
        if (down) {
            k32 = cvtW(pdc, 4 * planes, C);  fold(bdp, 4 * planes);
            conv(buf[curB], WA, nullptr, tD, C, H, W, 4 * planes, 1, 1, stride, 0, 1, 0,
                 k32, hd, wd);
            res = tD;
        }
        k32 = cvtW(pc3, 4 * planes, planes);  fold(b3p, 4 * planes);
        conv(t2, WA, res, tY, planes, h2, w2, 4 * planes, 1, 1, 1, 0, 1, 1, k32, h3, w3);

        curB = picks[3];
        C = 4 * planes; H = h3; W = w3;
    };

    // layer1: 64 planes x3 (stride 1, first has downsample)
    bottleneck(64, 1, true);
    bottleneck(64, 1, false);
    bottleneck(64, 1, false);
    // layer2: 128 planes x4 (first stride 2)
    bottleneck(128, 2, true);
    for (int i = 0; i < 3; ++i) bottleneck(128, 1, false);
    // layer3: 256 planes x23 (stride 1, first has downsample)
    bottleneck(256, 1, true);
    for (int i = 0; i < 22; ++i) bottleneck(256, 1, false);

    // --- backend: dilated 3x3 convs + relu ---
    int bc[7] = {1024, 512, 512, 512, 256, 128, 64};
    for (int i = 0; i < 6; ++i) {
        const float* w = next();
        const float* b = next();
        int k32 = cvtW(w, bc[i + 1], bc[i] * 9);
        biasOnly(b, bc[i + 1]);
        int nb = (curB + 1) % 5;
        int ho, wo;
        conv(buf[curB], WA, nullptr, buf[nb], bc[i], H, W, bc[i + 1],
             3, 3, 1, 2, 2, 1, k32, ho, wo);
        curB = nb; C = bc[i + 1]; H = ho; W = wo;   // stays 48x48
    }

    // --- convDU / convLR recurrent scans ---
    const float* duw = next(); const float* dub = next();
    const float* lrw = next(); const float* lrb = next();
    const float* ow  = next(); const float* ob  = next();
    k_cvt_pad<<<dim3((64 * 576 + 255) / 256), dim3(256), 0, stream>>>(
        duw, duW, 64, 576, 576);
    k_cvt_pad<<<dim3((64 * 576 + 255) / 256), dim3(256), 0, stream>>>(
        lrw, lrW, 64, 576, 576);

    u16* Y = buf[curB];            // [64, 48, 48] bf16, channel stride 2304
    dim3 sg(4);
    // rows: forward then backward, conv along w (element stride 1)
    for (int h = 1; h < 48; ++h)
        k_scan_step<<<sg, dim3(32), 0, stream>>>(Y + (h - 1) * 48, Y + h * 48,
                                                 duW, dub, 1, 2304, 48);
    for (int h = 46; h >= 0; --h)
        k_scan_step<<<sg, dim3(32), 0, stream>>>(Y + (h + 1) * 48, Y + h * 48,
                                                 duW, dub, 1, 2304, 48);
    // cols: forward then backward, conv along h (element stride 48)
    for (int xp = 1; xp < 48; ++xp)
        k_scan_step<<<sg, dim3(32), 0, stream>>>(Y + (xp - 1), Y + xp,
                                                 lrW, lrb, 48, 2304, 48);
    for (int xp = 46; xp >= 0; --xp)
        k_scan_step<<<sg, dim3(32), 0, stream>>>(Y + (xp + 1), Y + xp,
                                                 lrW, lrb, 48, 2304, 48);

    // --- output conv + relu + x8 nearest upsample ---
    k_out_up<<<dim3((2304 + 255) / 256), dim3(256), 0, stream>>>(
        Y, ow, ob, (float*)d_out);
}
